// TGN_58995670778162
// MI455X (gfx1250) — compile-verified, matrix operand order checked by reading, not run on hardware
//
#include <hip/hip_runtime.h>
#include <hip/hip_bf16.h>
#include <math.h>

// ---------------- problem constants ----------------
#define NNODES   100000
#define NEDGES   10000
#define DMEM     512
#define DEDGE    172
#define DRAW     1708            // 2*DMEM + DEDGE + DMEM
#define DHID     854
#define NSAMP    4000
#define MBSZ     200
#define NUMMB    20
#define ROWS     400             // 2*MBSZ events per step
#define G3       1536            // 3*DMEM

// ---------------- padded tile dims -----------------
#define MP       448             // rows padded: 400 -> 448 (7 * 64)
#define KP1      1728            // raw K padded: 1708 -> 1728 (54 * 32)
#define NP1      896             // hidden padded: 854 -> 896 (7 * 128)

typedef __attribute__((ext_vector_type(16))) __bf16 bf16x16;
typedef __attribute__((ext_vector_type(8)))  __bf16 bf16x8;
typedef __attribute__((ext_vector_type(8)))  float  f32x8;

// Load a 16x32 bf16 A/B WMMA fragment slice for this lane:
// element j (0..15) corresponds to K = 16*(j>>3) + 8*(lane>>4) + (j&7),
// i.e. two contiguous runs of 8 bf16 -> two 16B loads.
static __device__ inline bf16x16 load_frag(const __bf16* p) {
    bf16x8 lo = *(const bf16x8*)(p);
    bf16x8 hi = *(const bf16x8*)(p + 16);
    bf16x16 r;
#pragma unroll
    for (int j = 0; j < 8; ++j) { r[j] = lo[j]; r[j + 8] = hi[j]; }
    return r;
}

// ---------------------------------------------------------------------------
// Generic WMMA GEMM:  C[M x N] = A[M x K] * B[K x N] + bias,  B given
// transposed (BT[n*ldb + k]).  All dims padded; no bounds checks.
// Block: 256 threads (8 waves). Wave computes a 32x32 tile (2x2 WMMA tiles).
// Block tile: 64 (M) x 128 (N).  grid = (N/128, M/64).
// MODE 0: bf16 out, bias+relu.  MODE 1: bf16 out, bias.  MODE 2: f32 out, bias.
// ---------------------------------------------------------------------------
template <int MODE>
__global__ __launch_bounds__(256) void gemm_wmma(
    const __bf16* __restrict__ A, int lda,
    const __bf16* __restrict__ BT, int ldb,
    const float* __restrict__ bias,
    void* __restrict__ Cout, int ldc, int K)
{
    const int lane = threadIdx.x & 31;
    const int wave = threadIdx.x >> 5;
    const int wm   = wave & 1;
    const int wn   = wave >> 1;
    const int M0   = blockIdx.y * 64 + wm * 32;
    const int N0   = blockIdx.x * 128 + wn * 32;
    const int m    = lane & 15;       // row-of-A / col-of-B within tile
    const int half = lane >> 4;       // K-half select

    f32x8 acc00 = {}, acc01 = {}, acc10 = {}, acc11 = {};

    const __bf16* Ar0 = A  + (size_t)(M0 + m) * lda + 8 * half;
    const __bf16* Ar1 = Ar0 + (size_t)16 * lda;
    const __bf16* Br0 = BT + (size_t)(N0 + m) * ldb + 8 * half;
    const __bf16* Br1 = Br0 + (size_t)16 * ldb;

    for (int k = 0; k < K; k += 32) {
        bf16x16 a0 = load_frag(Ar0 + k);
        bf16x16 a1 = load_frag(Ar1 + k);
        bf16x16 b0 = load_frag(Br0 + k);
        bf16x16 b1 = load_frag(Br1 + k);
        acc00 = __builtin_amdgcn_wmma_f32_16x16x32_bf16(false, a0, false, b0, (short)0, acc00, false, false);
        acc01 = __builtin_amdgcn_wmma_f32_16x16x32_bf16(false, a0, false, b1, (short)0, acc01, false, false);
        acc10 = __builtin_amdgcn_wmma_f32_16x16x32_bf16(false, a1, false, b0, (short)0, acc10, false, false);
        acc11 = __builtin_amdgcn_wmma_f32_16x16x32_bf16(false, a1, false, b1, (short)0, acc11, false, false);
    }

    // Epilogue. C/D layout: lane gives N = (lane&15), VGPR r gives M = r + 8*half.
    f32x8 accs[2][2] = { { acc00, acc01 }, { acc10, acc11 } };
#pragma unroll
    for (int mi = 0; mi < 2; ++mi) {
#pragma unroll
        for (int ni = 0; ni < 2; ++ni) {
            const int Ncol = N0 + ni * 16 + m;
            const float bv = bias[Ncol];
#pragma unroll
            for (int r = 0; r < 8; ++r) {
                const int Mrow = M0 + mi * 16 + 8 * half + r;
                float v = accs[mi][ni][r] + bv;
                if (MODE == 0) v = fmaxf(v, 0.0f);
                if (MODE == 2) {
                    ((float*)Cout)[(size_t)Mrow * ldc + Ncol] = v;
                } else {
                    ((__bf16*)Cout)[(size_t)Mrow * ldc + Ncol] = (__bf16)v;
                }
            }
        }
    }
}

// ---------------------------------------------------------------------------
// Per-step gather: build raw[i] = [mem_s | mem_p | edge_feat | cos(dt*w+b)]
// (bf16, row stride KP1) and fp32 copy of mem_s for the GRU blend.
// One block per event row.
// ---------------------------------------------------------------------------
__global__ __launch_bounds__(256) void gather_build(
    const int* __restrict__ src, const int* __restrict__ dst,
    const float* __restrict__ edge_times, const int* __restrict__ edge_idxs,
    const float* __restrict__ memory, const float* __restrict__ lu,
    const float* __restrict__ edge_feats,
    const float* __restrict__ time_w, const float* __restrict__ time_b,
    __bf16* __restrict__ raw, float* __restrict__ mems, int step)
{
    const int i = blockIdx.x;                 // 0..399
    const int t = threadIdx.x;
    const int e = (i < MBSZ) ? i : (i - MBSZ);
    const int base = step * MBSZ + e;
    int node, partner;
    if (i < MBSZ) { node = src[base]; partner = dst[base]; }
    else          { node = dst[base]; partner = src[base]; }
    const float tt = edge_times[base];
    const int   ed = edge_idxs[base];

    const float* ms = memory + (size_t)node * DMEM;
    const float* mp = memory + (size_t)partner * DMEM;
    __bf16* r = raw + (size_t)i * KP1;
    const float dt = tt - lu[node];

    for (int c = t; c < DMEM; c += 256) {
        float v = ms[c];
        r[c] = (__bf16)v;
        mems[(size_t)i * DMEM + c] = v;
        r[DMEM + c] = (__bf16)mp[c];
        float te = cosf(dt * time_w[c] + time_b[c]);
        r[2 * DMEM + DEDGE + c] = (__bf16)te;
    }
    for (int c = t; c < DEDGE; c += 256)
        r[2 * DMEM + c] = (__bf16)edge_feats[(size_t)ed * DEDGE + c];
}

// ---------------------------------------------------------------------------
// GRU gates + "last writer wins" scatter. One block per event row.
// gx/gh already include b_ih/b_hh from the GEMM epilogue bias.
// ---------------------------------------------------------------------------
__global__ __launch_bounds__(256) void gru_gates_scatter(
    const int* __restrict__ src, const int* __restrict__ dst,
    const float* __restrict__ edge_times,
    const float* __restrict__ gx, const float* __restrict__ gh,
    const float* __restrict__ mems,
    float* __restrict__ memory, float* __restrict__ lu, int step)
{
    __shared__ int sNotWinner;
    const int i = blockIdx.x;                 // 0..399
    const int t = threadIdx.x;
    if (t == 0) sNotWinner = 0;
    __syncthreads();

    const int e = (i < MBSZ) ? i : (i - MBSZ);
    const int base = step * MBSZ;
    const int node = (i < MBSZ) ? src[base + e] : dst[base + e];

    // last occurrence of `node` in idx order wins
    for (int j = i + 1 + t; j < ROWS; j += 256) {
        int nj = (j < MBSZ) ? src[base + j] : dst[base + j - MBSZ];
        if (nj == node) sNotWinner = 1;
    }
    __syncthreads();
    if (sNotWinner) return;

    const float tt = edge_times[base + e];
    const float* gxr = gx + (size_t)i * G3;
    const float* ghr = gh + (size_t)i * G3;
    const float* msr = mems + (size_t)i * DMEM;
    float* mrow = memory + (size_t)node * DMEM;

    for (int c = t; c < DMEM; c += 256) {
        float rg = 1.0f / (1.0f + expf(-(gxr[c] + ghr[c])));
        float zg = 1.0f / (1.0f + expf(-(gxr[c + DMEM] + ghr[c + DMEM])));
        float ng = tanhf(gxr[c + 2 * DMEM] + rg * ghr[c + 2 * DMEM]);
        mrow[c] = (1.0f - zg) * ng + zg * msr[c];
    }
    if (t == 0) lu[node] = tt;
}

// ---------------------------------------------------------------------------
// One-time per call: transpose fp32 weight [K x N] -> bf16 [Npad x Kpad]
// (zero-padded), so WMMA B-fragments become contiguous 16B loads.
// ---------------------------------------------------------------------------
__global__ __launch_bounds__(256) void transpose_to_bf16(
    const float* __restrict__ srcW, __bf16* __restrict__ dstW,
    int K, int N, int Kpad, int Npad)
{
    size_t idx = (size_t)blockIdx.x * 256 + threadIdx.x;
    size_t total = (size_t)Kpad * Npad;
    if (idx >= total) return;
    int k = (int)(idx % Kpad);
    int n = (int)(idx / Kpad);
    float v = (k < K && n < N) ? srcW[(size_t)k * N + n] : 0.0f;
    dstW[idx] = (__bf16)v;
}

__global__ __launch_bounds__(256) void pad_bias(
    const float* __restrict__ b, float* __restrict__ bp, int N, int Npad)
{
    int i = blockIdx.x * 256 + threadIdx.x;
    if (i < Npad) bp[i] = (i < N) ? b[i] : 0.0f;
}

__global__ __launch_bounds__(256) void zero_u32(unsigned int* __restrict__ p, size_t n)
{
    size_t i = (size_t)blockIdx.x * 256 + threadIdx.x;
    if (i < n) p[i] = 0u;
}

// Final embeddings: out[k][i][:] = memory[n] + node_feats[n]
__global__ __launch_bounds__(256) void embed_out(
    const int* __restrict__ src, const int* __restrict__ dst,
    const int* __restrict__ neg,
    const float* __restrict__ memory, const float* __restrict__ node_feats,
    float* __restrict__ out)
{
    const int row = blockIdx.x;               // 0 .. 3*NSAMP-1
    const int k = row / NSAMP, i = row % NSAMP;
    const int n = (k == 0) ? src[i] : (k == 1) ? dst[i] : neg[i];
    const float* m = memory + (size_t)n * DMEM;
    const float* f = node_feats + (size_t)n * DMEM;
    float* o = out + (size_t)row * DMEM;
    for (int c = threadIdx.x; c < DMEM; c += 256) o[c] = m[c] + f[c];
}

// ---------------------------------------------------------------------------
extern "C" void kernel_launch(void* const* d_in, const int* in_sizes, int n_in,
                              void* d_out, int out_size, void* d_ws, size_t ws_size,
                              hipStream_t stream)
{
    (void)in_sizes; (void)n_in; (void)out_size; (void)ws_size;
    const int*   src        = (const int*)d_in[0];
    const int*   dst        = (const int*)d_in[1];
    const int*   neg        = (const int*)d_in[2];
    const float* edge_times = (const float*)d_in[3];
    const int*   edge_idxs  = (const int*)d_in[4];
    const float* memory0    = (const float*)d_in[5];
    const float* lu0        = (const float*)d_in[6];
    const float* node_feats = (const float*)d_in[7];
    const float* edge_feats = (const float*)d_in[8];
    const float* time_w     = (const float*)d_in[9];
    const float* time_b     = (const float*)d_in[10];
    const float* W1         = (const float*)d_in[11];
    const float* b1         = (const float*)d_in[12];
    const float* W2         = (const float*)d_in[13];
    const float* b2         = (const float*)d_in[14];
    const float* W_ih       = (const float*)d_in[15];
    const float* W_hh       = (const float*)d_in[16];
    const float* b_ih       = (const float*)d_in[17];
    const float* b_hh       = (const float*)d_in[18];

    // ---- carve workspace (256B aligned blocks) ----
    char* p = (char*)d_ws;
    auto carve = [&](size_t bytes) -> char* {
        char* r = p; p += (bytes + 255) & ~(size_t)255; return r;
    };
    float*  memory = (float*) carve((size_t)NNODES * DMEM * 4);
    float*  lu     = (float*) carve((size_t)NNODES * 4);
    __bf16* W1T    = (__bf16*)carve((size_t)NP1 * KP1 * 2);
    __bf16* W2T    = (__bf16*)carve((size_t)DMEM * NP1 * 2);
    __bf16* WihT   = (__bf16*)carve((size_t)G3 * DMEM * 2);
    __bf16* WhhT   = (__bf16*)carve((size_t)G3 * DMEM * 2);
    float*  b1p    = (float*) carve((size_t)NP1 * 4);
    __bf16* raw    = (__bf16*)carve((size_t)MP * KP1 * 2);
    __bf16* H      = (__bf16*)carve((size_t)MP * NP1 * 2);
    __bf16* msg    = (__bf16*)carve((size_t)MP * DMEM * 2);
    float*  mems   = (float*) carve((size_t)ROWS * DMEM * 4);
    float*  gx     = (float*) carve((size_t)MP * G3 * 4);
    float*  gh     = (float*) carve((size_t)MP * G3 * 4);

    // ---- init state (copy, never mutate inputs) ----
    hipMemcpyAsync(memory, memory0, (size_t)NNODES * DMEM * 4,
                   hipMemcpyDeviceToDevice, stream);
    hipMemcpyAsync(lu, lu0, (size_t)NNODES * 4, hipMemcpyDeviceToDevice, stream);

    // ---- one-time weight transposes (bf16, padded) ----
    {
        size_t n1 = (size_t)NP1 * KP1;
        transpose_to_bf16<<<(unsigned)((n1 + 255) / 256), 256, 0, stream>>>(
            W1, W1T, DRAW, DHID, KP1, NP1);
        size_t n2 = (size_t)DMEM * NP1;
        transpose_to_bf16<<<(unsigned)((n2 + 255) / 256), 256, 0, stream>>>(
            W2, W2T, DHID, DMEM, NP1, DMEM);
        size_t n3 = (size_t)G3 * DMEM;
        transpose_to_bf16<<<(unsigned)((n3 + 255) / 256), 256, 0, stream>>>(
            W_ih, WihT, DMEM, G3, DMEM, G3);
        transpose_to_bf16<<<(unsigned)((n3 + 255) / 256), 256, 0, stream>>>(
            W_hh, WhhT, DMEM, G3, DMEM, G3);
        pad_bias<<<(NP1 + 255) / 256, 256, 0, stream>>>(b1, b1p, DHID, NP1);
        // zero raw once: padding rows/cols stay zero across steps
        size_t nz = (size_t)MP * KP1 * 2 / 4;
        zero_u32<<<(unsigned)((nz + 255) / 256), 256, 0, stream>>>(
            (unsigned int*)raw, nz);
    }

    // ---- sequential scan over mini-batches ----
    for (int step = 0; step < NUMMB; ++step) {
        gather_build<<<ROWS, 256, 0, stream>>>(
            src, dst, edge_times, edge_idxs, memory, lu,
            edge_feats, time_w, time_b, raw, mems, step);

        // H = relu(raw @ W1 + b1)   [448 x 896], K = 1728
        gemm_wmma<0><<<dim3(NP1 / 128, MP / 64), 256, 0, stream>>>(
            raw, KP1, W1T, KP1, b1p, (void*)H, NP1, KP1);

        // msg = H @ W2 + b2         [448 x 512], K = 896
        gemm_wmma<1><<<dim3(DMEM / 128, MP / 64), 256, 0, stream>>>(
            H, NP1, W2T, NP1, b2, (void*)msg, DMEM, NP1);

        // gx = msg @ W_ih + b_ih    [448 x 1536], K = 512
        gemm_wmma<2><<<dim3(G3 / 128, MP / 64), 256, 0, stream>>>(
            msg, DMEM, WihT, DMEM, b_ih, (void*)gx, G3, DMEM);

        // gh = mem_s @ W_hh + b_hh  (mem_s bf16 = raw[:, 0:512]), K = 512
        gemm_wmma<2><<<dim3(G3 / 128, MP / 64), 256, 0, stream>>>(
            raw, KP1, WhhT, DMEM, b_hh, (void*)gh, G3, DMEM);

        gru_gates_scatter<<<ROWS, 256, 0, stream>>>(
            src, dst, edge_times, gx, gh, mems, memory, lu, step);
    }

    // ---- final embeddings ----
    embed_out<<<3 * NSAMP, 256, 0, stream>>>(
        src, dst, neg, memory, node_feats, (float*)d_out);
}
